// HiveMindGNN_73323681677408
// MI455X (gfx1250) — matrix operand order repaired and checked by reading, not compiled, not run on hardware
//
#include <hip/hip_runtime.h>
#include <hip/hip_bf16.h>

// ---------------------------------------------------------------------------
// CDNA5 (gfx1250) implementation of the HiveMind GNN reference.
// All 128x128 projections run through v_wmma_f32_16x16x32_bf16 (f32 accum).
// Irregular edge work (segment softmax, aggregation) uses atomics + wave
// reductions. wave32 assumed throughout (gfx1250).
// ---------------------------------------------------------------------------

typedef __attribute__((ext_vector_type(16))) __bf16 v16bf;
typedef __attribute__((ext_vector_type(8)))  float  v8f;

__device__ __forceinline__ unsigned short f2bf(float x) {
    unsigned int u = __float_as_uint(x);
    u = u + 0x7FFFu + ((u >> 16) & 1u);          // round-to-nearest-even
    return (unsigned short)(u >> 16);
}

// monotonic float <-> uint key for atomicMax-based segment max
__device__ __forceinline__ unsigned int fkey(float f) {
    unsigned int b = __float_as_uint(f);
    unsigned int mask = (unsigned int)(-(int)(b >> 31)) | 0x80000000u;
    return b ^ mask;
}
__device__ __forceinline__ float keyf(unsigned int u) {
    unsigned int mask = (u >> 31) ? 0x80000000u : 0xFFFFFFFFu;
    return __uint_as_float(u ^ mask);
}

// ---------------------------------------------------------------------------
// Weight pre-pack: W (128x128 f32 row-major, C = A@W) -> bf16 fragments in
// WMMA B-operand order. B layout (16-bit, 32x16 tile): lanes 0-15 hold
// K=0..7,16..23 (VGPR0-7, 2/VGPR), lanes 16-31 hold K=8..15,24..31; lane%16 = column n.
// Wp[((t*4 + kc)*32 + lane)*16 + j] = bf16(W[(kc*32 + (j>>3)*16 + (lane>>4)*8 + (j&7))][t*16 + (lane&15)])
// ---------------------------------------------------------------------------
__global__ void pack_w(const float* __restrict__ W, unsigned short* __restrict__ Wp) {
    int idx = blockIdx.x * blockDim.x + threadIdx.x;       // 0..16383
    if (idx >= 16384) return;
    int j    = idx & 15;
    int lane = (idx >> 4) & 31;
    int kc   = (idx >> 9) & 3;
    int t    = idx >> 11;
    int n = t * 16 + (lane & 15);
    int k = kc * 32 + (((j >> 3) << 4) + ((lane >> 4) << 3) + (j & 7));
    Wp[idx] = f2bf(W[k * 128 + n]);
}

// ---------------------------------------------------------------------------
// Generic GEMM: C[M x 128] = A[M x 128] @ W[128 x 128]  (bf16 in, f32 out)
// 128 threads = 4 waves; wave w computes column tiles 2w, 2w+1.
// 64 rows per block (4 x 16-row WMMA passes).
// A-fragment LDS layout (16-bit A 16x32): lane = ((kk>>3)&1)*16 + m,
// j = (kk>>4)*8 + (kk&7), kk = k % 32.
// Full-tile fast path: N=50000 and E=800000 are multiples of 16, so in
// practice every pass takes the unguarded path (one uniform branch, no
// per-row exec-mask churn in the epilogue).
// ---------------------------------------------------------------------------
__global__ void gemm128(const float* __restrict__ A,
                        const unsigned short* __restrict__ Wp,
                        float* __restrict__ C, int M) {
    __shared__ __align__(32) unsigned short As[4 * 32 * 16];   // 4KB
    const int tid  = threadIdx.x;
    const int wave = tid >> 5;
    const int lane = tid & 31;
    const int nCol = lane & 15;
    const int mRow = (lane >> 4) << 3;

    for (int it = 0; it < 4; ++it) {
        int rowBase = blockIdx.x * 64 + it * 16;
        if (rowBase >= M) break;                       // uniform per block
        const bool full = (rowBase + 16 <= M);         // uniform per block
        // stage 16x128 A tile into fragment layout (float4 coalesced loads)
        const float4* A4 = reinterpret_cast<const float4*>(A) + (size_t)rowBase * 32;
        // prefetch next 16-row A tile while this pass computes
        if (rowBase + 16 + (tid >> 3) < M)
            __builtin_prefetch(A + (size_t)(rowBase + 16 + (tid >> 3)) * 128 + (tid & 7) * 16, 0, 1);
        if (full) {
            #pragma unroll
            for (int i = 0; i < 4; ++i) {
                int q4 = tid * 4 + i;                  // 0..511
                int m  = q4 >> 5;
                float4 av = A4[q4];
                float vals[4] = {av.x, av.y, av.z, av.w};
                int kbase = (q4 & 31) * 4;
                #pragma unroll
                for (int u = 0; u < 4; ++u) {
                    int k  = kbase + u;
                    int kc = k >> 5, kk = k & 31;
                    int lgrp = (kk >> 3) & 1;
                    int j = ((kk >> 4) << 3) | (kk & 7);
                    As[(kc * 32 + (lgrp * 16 + m)) * 16 + j] = f2bf(vals[u]);
                }
            }
        } else {
            for (int i = 0; i < 4; ++i) {
                int q4 = tid * 4 + i;
                int m  = q4 >> 5;
                int gr = rowBase + m;
                float4 av = {0.f, 0.f, 0.f, 0.f};
                if (gr < M) av = A4[q4];
                float vals[4] = {av.x, av.y, av.z, av.w};
                int kbase = (q4 & 31) * 4;
                #pragma unroll
                for (int u = 0; u < 4; ++u) {
                    int k  = kbase + u;
                    int kc = k >> 5, kk = k & 31;
                    int lgrp = (kk >> 3) & 1;
                    int j = ((kk >> 4) << 3) | (kk & 7);
                    As[(kc * 32 + (lgrp * 16 + m)) * 16 + j] = f2bf(vals[u]);
                }
            }
        }
        __syncthreads();

        v8f acc0 = {0.f, 0.f, 0.f, 0.f, 0.f, 0.f, 0.f, 0.f};
        v8f acc1 = {0.f, 0.f, 0.f, 0.f, 0.f, 0.f, 0.f, 0.f};
        #pragma unroll
        for (int kc = 0; kc < 4; ++kc) {
            v16bf a  = *reinterpret_cast<const v16bf*>(&As[(kc * 32 + lane) * 16]);
            v16bf b0 = *reinterpret_cast<const v16bf*>(
                &Wp[(((wave * 2 + 0) * 4 + kc) * 32 + lane) * 16]);
            v16bf b1 = *reinterpret_cast<const v16bf*>(
                &Wp[(((wave * 2 + 1) * 4 + kc) * 32 + lane) * 16]);
            acc0 = __builtin_amdgcn_wmma_f32_16x16x32_bf16(
                false, a, false, b0, (short)0, acc0, false, false);
            acc1 = __builtin_amdgcn_wmma_f32_16x16x32_bf16(
                false, a, false, b1, (short)0, acc1, false, false);
        }
        // C/D layout: lane = (m/8)*16 + n, vgpr r -> m = (lane>>4)*8 + r
        if (full) {
            float* Cb = C + (size_t)(rowBase + mRow) * 128 + wave * 32 + nCol;
            #pragma unroll
            for (int r = 0; r < 8; ++r) {
                Cb[(size_t)r * 128]      = acc0[r];
                Cb[(size_t)r * 128 + 16] = acc1[r];
            }
        } else {
            #pragma unroll
            for (int r = 0; r < 8; ++r) {
                int gm = rowBase + mRow + r;
                if (gm < M) {
                    C[(size_t)gm * 128 + wave * 32 + nCol]      = acc0[r];
                    C[(size_t)gm * 128 + wave * 32 + nCol + 16] = acc1[r];
                }
            }
        }
        __syncthreads();
    }
}

// ---------------------------------------------------------------------------
// Node encoder stage 1: h1 = relu(nf @ Wn1 + bn1)   (K = 7, scalar)
// ---------------------------------------------------------------------------
__global__ void node_enc1_k(const float* __restrict__ nf, const float* __restrict__ Wn1,
                            const float* __restrict__ bn1, float* __restrict__ h1, int N) {
    size_t idx = (size_t)blockIdx.x * blockDim.x + threadIdx.x;
    if (idx >= (size_t)N * 128) return;
    int n = (int)(idx >> 7), c = (int)(idx & 127);
    float acc = bn1[c];
    #pragma unroll
    for (int f = 0; f < 7; ++f) acc += nf[(size_t)n * 7 + f] * Wn1[f * 128 + c];
    h1[idx] = fmaxf(acc, 0.f);
}

// edge encoder: ee = relu(ea @ We1 + be1)  (K = 2, scalar)
__global__ void edge_enc_k(const float* __restrict__ ea, const float* __restrict__ We1,
                           const float* __restrict__ be1, float* __restrict__ ee, int E) {
    size_t idx = (size_t)blockIdx.x * blockDim.x + threadIdx.x;
    if (idx >= (size_t)E * 128) return;
    int e = (int)(idx >> 7), c = (int)(idx & 127);
    float acc = ea[(size_t)e * 2] * We1[c] + ea[(size_t)e * 2 + 1] * We1[128 + c] + be1[c];
    ee[idx] = fmaxf(acc, 0.f);
}

// in-degree (float) via atomics
__global__ void deg_k(const int* __restrict__ dst, float* __restrict__ degf, int E) {
    int e = blockIdx.x * blockDim.x + threadIdx.x;
    if (e < E) atomicAdd(&degf[dst[e]], 1.0f);
}

// sinusoidal positional encoding added to x
__global__ void posenc_k(const float* __restrict__ degf, float* __restrict__ x, int N) {
    int idx = blockIdx.x * blockDim.x + threadIdx.x;
    if (idx >= N * 64) return;
    int n = idx >> 6, i = idx & 63;
    float freq = __expf(-9.210340371976184f * (float)i / 64.f);   // ln(10000)
    float ang = degf[n] * freq;
    x[(size_t)n * 128 + 2 * i]     += __sinf(ang);
    x[(size_t)n * 128 + 2 * i + 1] += __cosf(ang);
}

// ---------------------------------------------------------------------------
// Row LayerNorm (wave per row): Out = LN(X [+ bias] [+ Res]) * g + b
// ---------------------------------------------------------------------------
__global__ void ln_rows_k(const float* __restrict__ X, const float* __restrict__ Res,
                          const float* __restrict__ bias, const float* __restrict__ g,
                          const float* __restrict__ b, float* __restrict__ Out, int M) {
    int row = blockIdx.x * (blockDim.x >> 5) + (threadIdx.x >> 5);
    if (row >= M) return;
    int lane = threadIdx.x & 31;
    float v[4], s = 0.f, s2 = 0.f;
    #pragma unroll
    for (int i = 0; i < 4; ++i) {
        int c = i * 32 + lane;
        float t = X[(size_t)row * 128 + c];
        if (bias) t += bias[c];
        if (Res)  t += Res[(size_t)row * 128 + c];
        v[i] = t; s += t; s2 += t * t;
    }
    for (int o = 16; o > 0; o >>= 1) {
        s  += __shfl_xor(s,  o, 32);
        s2 += __shfl_xor(s2, o, 32);
    }
    float mean = s * (1.f / 128.f);
    float var  = s2 * (1.f / 128.f) - mean * mean;
    float rs = rsqrtf(var + 1e-5f);
    #pragma unroll
    for (int i = 0; i < 4; ++i) {
        int c = i * 32 + lane;
        Out[(size_t)row * 128 + c] = (v[i] - mean) * rs * g[c] + b[c];
    }
}

// ---------------------------------------------------------------------------
// Attention: per-edge logits (wave per edge, 4 heads x DH=32) + segment max
// ---------------------------------------------------------------------------
__global__ void attn_logits_k(const float* __restrict__ q, const float* __restrict__ k,
                              const float* __restrict__ ek, const int* __restrict__ src,
                              const int* __restrict__ dst, float* __restrict__ logit,
                              unsigned int* __restrict__ mkey, int E) {
    int e = blockIdx.x * (blockDim.x >> 5) + (threadIdx.x >> 5);
    if (e >= E) return;
    int lane = threadIdx.x & 31;
    int s = src[e], d = dst[e];
    const float* qd = q  + (size_t)d * 128;
    const float* ks = k  + (size_t)s * 128;
    const float* ee = ek + (size_t)e * 128;
    float l0 = qd[lane]      * (ks[lane]      + ee[lane]);
    float l1 = qd[lane + 32] * (ks[lane + 32] + ee[lane + 32]);
    float l2 = qd[lane + 64] * (ks[lane + 64] + ee[lane + 64]);
    float l3 = qd[lane + 96] * (ks[lane + 96] + ee[lane + 96]);
    for (int o = 16; o > 0; o >>= 1) {
        l0 += __shfl_xor(l0, o, 32); l1 += __shfl_xor(l1, o, 32);
        l2 += __shfl_xor(l2, o, 32); l3 += __shfl_xor(l3, o, 32);
    }
    if (lane == 0) {
        const float sc = 0.17677669529663687f;   // 1/sqrt(32)
        float L[4] = {l0 * sc, l1 * sc, l2 * sc, l3 * sc};
        #pragma unroll
        for (int h = 0; h < 4; ++h) {
            logit[(size_t)e * 4 + h] = L[h];
            atomicMax(&mkey[(size_t)d * 4 + h], fkey(L[h]));
        }
    }
}

// exp(logit - m) in place + segment sum
__global__ void attn_exp_k(float* __restrict__ la, const unsigned int* __restrict__ mkey,
                           const int* __restrict__ dst, float* __restrict__ ssum, int E) {
    size_t idx = (size_t)blockIdx.x * blockDim.x + threadIdx.x;
    if (idx >= (size_t)E * 4) return;
    int e = (int)(idx >> 2), h = (int)(idx & 3);
    int d = dst[e];
    float m = keyf(mkey[(size_t)d * 4 + h]);
    float a = __expf(la[idx] - m);
    la[idx] = a;
    atomicAdd(&ssum[(size_t)d * 4 + h], a);
}

// agg[dst] += alpha * v[src]
__global__ void attn_agg_k(const float* __restrict__ alpha, const float* __restrict__ ssum,
                           const float* __restrict__ v, const int* __restrict__ src,
                           const int* __restrict__ dst, float* __restrict__ agg, int E) {
    size_t idx = (size_t)blockIdx.x * blockDim.x + threadIdx.x;
    if (idx >= (size_t)E * 128) return;
    int e = (int)(idx >> 7), c = (int)(idx & 127), h = c >> 5;
    int s = src[e], d = dst[e];
    float a = alpha[(size_t)e * 4 + h] / (ssum[(size_t)d * 4 + h] + 1e-9f);
    atomicAdd(&agg[(size_t)d * 128 + c], a * v[(size_t)s * 128 + c]);
}

// edge update: ee = LN(ee + relu(xa[src] + xb[dst] + ec))  (wave per edge)
__global__ void edge_update_k(const float* __restrict__ xa, const float* __restrict__ xb,
                              const float* __restrict__ ec, float* __restrict__ ee,
                              const int* __restrict__ src, const int* __restrict__ dst,
                              const float* __restrict__ g, const float* __restrict__ b, int E) {
    int e = blockIdx.x * (blockDim.x >> 5) + (threadIdx.x >> 5);
    if (e >= E) return;
    int lane = threadIdx.x & 31;
    int s = src[e], d = dst[e];
    float t[4], sm = 0.f, s2 = 0.f;
    #pragma unroll
    for (int i = 0; i < 4; ++i) {
        int c = i * 32 + lane;
        float hh = xa[(size_t)s * 128 + c] + xb[(size_t)d * 128 + c] + ec[(size_t)e * 128 + c];
        hh = fmaxf(hh, 0.f);
        float tv = ee[(size_t)e * 128 + c] + hh;
        t[i] = tv; sm += tv; s2 += tv * tv;
    }
    for (int o = 16; o > 0; o >>= 1) {
        sm += __shfl_xor(sm, o, 32);
        s2 += __shfl_xor(s2, o, 32);
    }
    float mean = sm * (1.f / 128.f);
    float var  = s2 * (1.f / 128.f) - mean * mean;
    float rs = rsqrtf(var + 1e-5f);
    #pragma unroll
    for (int i = 0; i < 4; ++i) {
        int c = i * 32 + lane;
        ee[(size_t)e * 128 + c] = (t[i] - mean) * rs * g[c] + b[c];
    }
}

// edge scorer (wave per edge)
__global__ void score_k(const float* __restrict__ xsa, const float* __restrict__ xsb,
                        const float* __restrict__ esc, const int* __restrict__ src,
                        const int* __restrict__ dst, const float* __restrict__ bs1,
                        const float* __restrict__ ws2, const float* __restrict__ bs2,
                        float* __restrict__ out, int E) {
    int e = blockIdx.x * (blockDim.x >> 5) + (threadIdx.x >> 5);
    if (e >= E) return;
    int lane = threadIdx.x & 31;
    int s = src[e], d = dst[e];
    float acc = 0.f;
    #pragma unroll
    for (int i = 0; i < 4; ++i) {
        int c = i * 32 + lane;
        float hh = xsa[(size_t)s * 128 + c] + xsb[(size_t)d * 128 + c] +
                   esc[(size_t)e * 128 + c] + bs1[c];
        acc += fmaxf(hh, 0.f) * ws2[c];
    }
    for (int o = 16; o > 0; o >>= 1) acc += __shfl_xor(acc, o, 32);
    if (lane == 0) out[e] = acc + bs2[0];
}

__global__ void copy_k(const float* __restrict__ src, float* __restrict__ dst, size_t n) {
    size_t i = (size_t)blockIdx.x * blockDim.x + threadIdx.x;
    if (i < n) dst[i] = src[i];
}

// ---------------------------------------------------------------------------
extern "C" void kernel_launch(void* const* d_in, const int* in_sizes, int n_in,
                              void* d_out, int out_size, void* d_ws, size_t ws_size,
                              hipStream_t stream) {
    (void)n_in; (void)out_size; (void)ws_size;
    const float* nf   = (const float*)d_in[0];
    const float* ea   = (const float*)d_in[1];
    const int*   ei   = (const int*)  d_in[2];
    const float* Wn1  = (const float*)d_in[3];
    const float* bn1  = (const float*)d_in[4];
    const float* Wn2  = (const float*)d_in[5];
    const float* bn2  = (const float*)d_in[6];
    const float* gn   = (const float*)d_in[7];
    const float* bnb  = (const float*)d_in[8];
    const float* We1  = (const float*)d_in[9];
    const float* be1  = (const float*)d_in[10];
    const float* Wq   = (const float*)d_in[11];
    const float* Wk   = (const float*)d_in[12];
    const float* Wv   = (const float*)d_in[13];
    const float* Wed  = (const float*)d_in[14];
    const float* Wo   = (const float*)d_in[15];
    const float* Wa   = (const float*)d_in[16];
    const float* Wb   = (const float*)d_in[17];
    const float* Wc   = (const float*)d_in[18];
    const float* g1   = (const float*)d_in[19];
    const float* b1   = (const float*)d_in[20];
    const float* g2   = (const float*)d_in[21];
    const float* b2   = (const float*)d_in[22];
    const float* Wsa  = (const float*)d_in[23];
    const float* Wsb  = (const float*)d_in[24];
    const float* Wsc  = (const float*)d_in[25];
    const float* bs1  = (const float*)d_in[26];
    const float* ws2  = (const float*)d_in[27];
    const float* bs2  = (const float*)d_in[28];

    const int N = in_sizes[0] / 7;
    const int E = in_sizes[1] / 2;
    const int* src = ei;
    const int* dst = ei + E;

    // ---- workspace bump allocator (256B aligned) ----
    char* base = (char*)d_ws;
    size_t off = 0;
    auto alloc = [&](size_t bytes) -> char* {
        off = (off + 255) & ~(size_t)255;
        char* p = base + off;
        off += bytes;
        return p;
    };
    unsigned short* wp = (unsigned short*)alloc((size_t)28 * 16384 * 2); // packed weights
    float* x    = (float*)alloc((size_t)N * 128 * 4);   // node state
    float* nb1  = (float*)alloc((size_t)N * 128 * 4);   // h1 / agg
    float* nb2  = (float*)alloc((size_t)N * 128 * 4);   // q / o / xsa
    float* nb3  = (float*)alloc((size_t)N * 128 * 4);   // k / xa / xsb
    float* nb4  = (float*)alloc((size_t)N * 128 * 4);   // v / xb
    float* ee   = (float*)alloc((size_t)E * 128 * 4);   // edge embedding
    float* ebuf = (float*)alloc((size_t)E * 128 * 4);   // ek / ec / esc
    float* alpha = (float*)alloc((size_t)E * 4 * 4);    // logits -> alpha
    unsigned int* mkey = (unsigned int*)alloc((size_t)N * 4 * 4);
    float* ssum = (float*)alloc((size_t)N * 4 * 4);
    float* degf = (float*)alloc((size_t)N * 4);

    // ---- pack all 128x128 weights to bf16 WMMA fragment layout ----
    const float* wsrc[28];
    wsrc[0] = Wn2;
    for (int l = 0; l < 3; ++l) {
        wsrc[1  + l] = Wq  + (size_t)l * 16384;
        wsrc[4  + l] = Wk  + (size_t)l * 16384;
        wsrc[7  + l] = Wv  + (size_t)l * 16384;
        wsrc[10 + l] = Wed + (size_t)l * 16384;
        wsrc[13 + l] = Wo  + (size_t)l * 16384;
        wsrc[16 + l] = Wa  + (size_t)l * 16384;
        wsrc[19 + l] = Wb  + (size_t)l * 16384;
        wsrc[22 + l] = Wc  + (size_t)l * 16384;
    }
    wsrc[25] = Wsa; wsrc[26] = Wsb; wsrc[27] = Wsc;
    for (int i = 0; i < 28; ++i)
        pack_w<<<64, 256, 0, stream>>>(wsrc[i], wp + (size_t)i * 16384);

    auto gemm = [&](const float* A, int wi, float* C, int M) {
        gemm128<<<(M + 63) / 64, 128, 0, stream>>>(A, wp + (size_t)wi * 16384, C, M);
    };
    auto ln = [&](const float* X, const float* Res, const float* bias,
                  const float* g, const float* b, float* Out, int M) {
        ln_rows_k<<<(M + 7) / 8, 256, 0, stream>>>(X, Res, bias, g, b, Out, M);
    };

    // ---- encoders ----
    node_enc1_k<<<(int)(((size_t)N * 128 + 255) / 256), 256, 0, stream>>>(nf, Wn1, bn1, nb1, N);
    gemm(nb1, 0, x, N);                               // x_pre = h1 @ Wn2
    ln(x, nullptr, bn2, gn, bnb, x, N);               // x = LN(x_pre + bn2)
    edge_enc_k<<<(int)(((size_t)E * 128 + 255) / 256), 256, 0, stream>>>(ea, We1, be1, ee, E);

    // ---- degree positional encoding ----
    hipMemsetAsync(degf, 0, (size_t)N * 4, stream);
    deg_k<<<(E + 255) / 256, 256, 0, stream>>>(dst, degf, E);
    posenc_k<<<(N * 64 + 255) / 256, 256, 0, stream>>>(degf, x, N);

    // ---- attention layers ----
    for (int l = 0; l < 3; ++l) {
        gemm(x, 1 + l, nb2, N);                       // q
        gemm(x, 4 + l, nb3, N);                       // k
        gemm(x, 7 + l, nb4, N);                       // v
        gemm(ee, 10 + l, ebuf, E);                    // ek = edge_emb @ Wed

        hipMemsetAsync(mkey, 0, (size_t)N * 16, stream);
        hipMemsetAsync(ssum, 0, (size_t)N * 16, stream);
        hipMemsetAsync(nb1, 0, (size_t)N * 128 * 4, stream);   // agg

        attn_logits_k<<<(E + 7) / 8, 256, 0, stream>>>(nb2, nb3, ebuf, src, dst, alpha, mkey, E);
        attn_exp_k<<<(int)(((size_t)E * 4 + 255) / 256), 256, 0, stream>>>(alpha, mkey, dst, ssum, E);
        attn_agg_k<<<(int)(((size_t)E * 128 + 255) / 256), 256, 0, stream>>>(
            alpha, ssum, nb4, src, dst, nb1, E);

        gemm(nb1, 13 + l, nb2, N);                    // o = agg @ Wo
        ln(nb2, x, nullptr, g1 + l * 128, b1 + l * 128, x, N);   // x = LN(x + o)

        gemm(x, 16 + l, nb3, N);                      // xa = x @ Wa
        gemm(x, 19 + l, nb4, N);                      // xb = x @ Wb
        gemm(ee, 22 + l, ebuf, E);                    // ec = edge_emb @ Wc
        edge_update_k<<<(E + 7) / 8, 256, 0, stream>>>(
            nb3, nb4, ebuf, ee, src, dst, g2 + l * 128, b2 + l * 128, E);
    }

    // ---- edge scorer ----
    gemm(x, 25, nb2, N);                              // xsa
    gemm(x, 26, nb3, N);                              // xsb
    gemm(ee, 27, ebuf, E);                            // esc
    float* out_x      = (float*)d_out;
    float* out_logits = out_x + (size_t)N * 128;
    score_k<<<(E + 7) / 8, 256, 0, stream>>>(nb2, nb3, ebuf, src, dst, bs1, ws2, bs2,
                                             out_logits, E);
    copy_k<<<(int)(((size_t)N * 128 + 255) / 256), 256, 0, stream>>>(x, out_x, (size_t)N * 128);
}
